// SelfAttention_14534169330105
// MI455X (gfx1250) — matrix operand order repaired and checked by reading, not compiled, not run on hardware
//
#include <hip/hip_runtime.h>

// Problem constants (from reference): x is [8, 256, 64, 64] fp32.
#define B_  8
#define C_  256
#define N_  4096
#define LOG2E 1.4426950408889634f

typedef __bf16 bf16;
typedef __attribute__((ext_vector_type(16))) __bf16 v16bf;
typedef __attribute__((ext_vector_type(8)))  __bf16 v8bf;
typedef __attribute__((ext_vector_type(4)))  __bf16 v4bf;
typedef __attribute__((ext_vector_type(8)))  float  v8f;

union V16 { v16bf v; v8bf h[2]; };

// LDS tile geometry (elements). Pitches padded for bank spread & 16B align.
#define KPITCH 40              // 32 keys + 8 pad  (80 B rows, bank stride 20)
#define VPITCH 272             // 256 c + 16 pad   (544 B rows, bank stride 8)
#define KSZ    (256 * KPITCH)  // one K-slab buffer: 256 c-rows x 32 keys
#define VSZ    (32 * VPITCH)   // one V-tile buffer: 32 keys x 256 c
#define PPITCH 40
#define SMEM_ELEMS (2 * KSZ + 2 * VSZ + 8 * 16 * PPITCH)
#define SMEM_BYTES (SMEM_ELEMS * 2)

// ---------------------------------------------------------------------------
// CDNA5 async Global->LDS DMA (ASYNCcnt-tracked), per CDNA5 ISA/08:
//   global_load_async_to_lds_b128 vdst(lds byte offset), vaddr(64b), off
// ---------------------------------------------------------------------------
__device__ __forceinline__ unsigned lds_off_of(const void* p) {
  return (unsigned)(unsigned long long)
      ((__attribute__((address_space(3))) const char*)p);
}
__device__ __forceinline__ void async_copy_b128(unsigned lds_byte_off,
                                                const void* gaddr) {
  asm volatile("global_load_async_to_lds_b128 %0, %1, off"
               :: "v"(lds_byte_off), "v"(gaddr) : "memory");
}
__device__ __forceinline__ void wait_async0() {
  asm volatile("s_wait_asynccnt 0x0" ::: "memory");
}

// ---------------------------------------------------------------------------
// Pass 1a: fp32 [B,C,N] -> bf16 [B,C,N]  (rows are rows of K^T: B-fragments)
// ---------------------------------------------------------------------------
__global__ void convert_bf16_kernel(const float* __restrict__ x,
                                    bf16* __restrict__ xc) {
  size_t i = ((size_t)blockIdx.x * blockDim.x + threadIdx.x) * 4;
  float4 f = *(const float4*)(x + i);
  v4bf o;
  o[0] = (bf16)f.x; o[1] = (bf16)f.y; o[2] = (bf16)f.z; o[3] = (bf16)f.w;
  *(v4bf*)(xc + i) = o;
}

// ---------------------------------------------------------------------------
// Pass 1b: fp32 [B,C,N] -> bf16 [B,N,C] transpose (Q A-fragments, V rows).
// ---------------------------------------------------------------------------
__global__ void transpose_bf16_kernel(const float* __restrict__ x,
                                      bf16* __restrict__ xb) {
  __shared__ float tile[32][33];
  const int b  = blockIdx.z;
  const int c0 = blockIdx.y * 32;
  const int n0 = blockIdx.x * 32;
  for (int i = threadIdx.y; i < 32; i += 8)
    tile[i][threadIdx.x] = x[((size_t)b * C_ + c0 + i) * N_ + n0 + threadIdx.x];
  __syncthreads();
  for (int i = threadIdx.y; i < 32; i += 8)
    xb[((size_t)b * N_ + n0 + i) * C_ + c0 + threadIdx.x] =
        (bf16)tile[threadIdx.x][i];
}

// ---------------------------------------------------------------------------
// Pass 2: flash attention. 8 waves per block; each wave owns 16 query rows,
// a 16x256 f32 accumulator, and the 8 Q A-fragments held in registers for
// the whole kernel. K/V tiles for each 32-key block are DMA'd into
// double-buffered LDS with global_load_async_to_lds_b128 issued one block
// ahead (overlapping the WMMA stream), closed by s_wait_asynccnt 0 + the
// workgroup barrier. Softmax denominators are accumulated by an extra WMMA
// against an all-ones B-fragment (C-layout, rescales with the output acc),
// removing the per-block row-sum shuffle reduction.
// ---------------------------------------------------------------------------
__launch_bounds__(256, 1)
__global__ void fa_kernel(const bf16* __restrict__ Xb,  // [B,N,C] (Q and V)
                          const bf16* __restrict__ Xc,  // [B,C,N] (K^T rows)
                          float* __restrict__ out) {    // [B,C,N]
  extern __shared__ __align__(16) bf16 smem[];
  bf16* KsB = smem;                       // [2][256][KPITCH]
  bf16* VsB = smem + 2 * KSZ;             // [2][32][VPITCH]
  bf16* Pw  = smem + 2 * KSZ + 2 * VSZ;   // [8][16][PPITCH]

  const int tid     = threadIdx.x;
  const int lane    = tid & 31;
  const int wave    = tid >> 5;
  const int b       = blockIdx.y;
  const int m0      = blockIdx.x * 128 + wave * 16;  // query-tile base row
  const int col     = lane & 15;                     // C/D fragment column
  const int halfOff = (lane >> 4) * 8;               // hi half -> rows/k +8

  const bf16* Qrow  = Xb + ((size_t)b * N_ + m0 + (lane & 15)) * C_;
  const bf16* Vbase = Xb + (size_t)b * N_ * C_;
  const bf16* Kbase = Xc + (size_t)b * C_ * N_;
  bf16* Pme = Pw + wave * 16 * PPITCH;

  // Per-thread DMA slice: 64B of the K-slab (row c=tid, 32 keys) and 64B of
  // the V-tile (row tid>>3, 32 c starting at (tid&7)*32).
  const int vRow = tid >> 3;
  const int vCol = (tid & 7) * 32;
  const unsigned ldsK0 = lds_off_of(KsB + (size_t)tid * KPITCH);
  const unsigned ldsV0 = lds_off_of(VsB + (size_t)vRow * VPITCH + vCol);

  const v8f vzero = {0.f, 0.f, 0.f, 0.f, 0.f, 0.f, 0.f, 0.f};
  v8f acc[16];
#pragma unroll
  for (int t = 0; t < 16; ++t) acc[t] = vzero;
  v8f lacc = vzero;  // softmax denominators, same C layout/rescale as acc
  float mrow[8];
#pragma unroll
  for (int r = 0; r < 8; ++r) mrow[r] = -__builtin_inff();

  // All-ones B-fragment: wmma(P, ones) accumulates row sums of P into lacc.
  v16bf vones;
#pragma unroll
  for (int i = 0; i < 16; ++i) vones[i] = (bf16)1.0f;

  // Q A-fragments resident in registers for the whole key loop (64 VGPRs).
  V16 qfrag[8];
#pragma unroll
  for (int k = 0; k < 8; ++k) {
    qfrag[k].h[0] = *(const v8bf*)(Qrow + k * 32 + halfOff);
    qfrag[k].h[1] = *(const v8bf*)(Qrow + k * 32 + 16 + halfOff);
  }

  // ---- stage block jb = 0 into buffer 0 via async DMA ----
#pragma unroll
  for (int i = 0; i < 4; ++i) {
    async_copy_b128(ldsK0 + i * 16, Kbase + (size_t)tid * N_ + i * 8);
    async_copy_b128(ldsV0 + i * 16, Vbase + (size_t)vRow * C_ + vCol + i * 8);
  }
  wait_async0();
  __syncthreads();

  int buf = 0;
  for (int jb = 0; jb < N_; jb += 32) {
    const bool more = (jb + 32) < N_;
    if (more) {  // kick off next block's DMA before computing this one
      const unsigned kOff = (unsigned)((buf ^ 1) * KSZ) * 2u;
      const unsigned vOff = (unsigned)((buf ^ 1) * VSZ) * 2u;
      const bf16* gK = Kbase + (size_t)tid * N_ + jb + 32;
      const bf16* gV = Vbase + (size_t)(jb + 32 + vRow) * C_ + vCol;
#pragma unroll
      for (int i = 0; i < 4; ++i) {
        async_copy_b128(ldsK0 + kOff + i * 16, gK + i * 8);
        async_copy_b128(ldsV0 + vOff + i * 16, gV + i * 8);
      }
      if (jb + 64 < N_) {  // warm L2 two blocks ahead for the DMA engine
        __builtin_prefetch(Kbase + (size_t)tid * N_ + jb + 64, 0, 0);
        __builtin_prefetch(Vbase + (size_t)(jb + 64 + vRow) * C_ + vCol, 0, 0);
      }
    }

    const bf16* Kcur = KsB + (size_t)buf * KSZ;
    const bf16* Vcur = VsB + (size_t)buf * VSZ;

    // ---- S = Q K^T : 16x32 score tile (two 16x16 C-fragments) ----
    v8f s0 = vzero, s1 = vzero;
#pragma unroll
    for (int k = 0; k < 8; ++k) {
      v16bf kb0, kb1;
#pragma unroll
      for (int s = 0; s < 16; ++s) {
        const int kk = (s < 8 ? s : s + 8) + halfOff;
        kb0[s] = Kcur[(size_t)(k * 32 + kk) * KPITCH + col];
        kb1[s] = Kcur[(size_t)(k * 32 + kk) * KPITCH + 16 + col];
      }
      s0 = __builtin_amdgcn_wmma_f32_16x16x32_bf16(false, qfrag[k].v, false,
                                                   kb0, (short)0, s0, false,
                                                   false);
      s1 = __builtin_amdgcn_wmma_f32_16x16x32_bf16(false, qfrag[k].v, false,
                                                   kb1, (short)0, s1, false,
                                                   false);
    }

    // ---- online softmax: row max (4-step shuffle tree), exp, rescale ----
    float p0[8], p1[8], scale[8];
    bool resc = false;
#pragma unroll
    for (int r = 0; r < 8; ++r) {
      float mx = fmaxf(s0[r], s1[r]);
#pragma unroll
      for (int off = 8; off >= 1; off >>= 1)
        mx = fmaxf(mx, __shfl_xor(mx, off, 16));
      const float mnew = fmaxf(mrow[r], mx);
      scale[r] = exp2f((mrow[r] - mnew) * LOG2E);
      resc |= (scale[r] != 1.0f);
      p0[r] = exp2f((s0[r] - mnew) * LOG2E);
      p1[r] = exp2f((s1[r] - mnew) * LOG2E);
      mrow[r] = mnew;
    }
    if (__any(resc)) {  // running max moved: rescale numerator + denominator
#pragma unroll
      for (int t = 0; t < 16; ++t) {
        v8f a = acc[t];
#pragma unroll
        for (int r = 0; r < 8; ++r) a[r] *= scale[r];
        acc[t] = a;
      }
#pragma unroll
      for (int r = 0; r < 8; ++r) lacc[r] *= scale[r];
    }

    // ---- C-layout P -> bf16 A-layout via per-wave LDS tile (same-wave DS
    //      ordering makes the store->load sequence safe without a barrier) --
#pragma unroll
    for (int r = 0; r < 8; ++r) {
      Pme[(r + halfOff) * PPITCH + col]      = (bf16)p0[r];
      Pme[(r + halfOff) * PPITCH + 16 + col] = (bf16)p1[r];
    }
    V16 pa;
    pa.h[0] = *(const v8bf*)(Pme + (lane & 15) * PPITCH + halfOff);
    pa.h[1] = *(const v8bf*)(Pme + (lane & 15) * PPITCH + 16 + halfOff);

    // ---- denominators: lacc += P * ones (row sums, matrix-unit reduce) ----
    lacc = __builtin_amdgcn_wmma_f32_16x16x32_bf16(false, pa.v, false, vones,
                                                   (short)0, lacc, false,
                                                   false);

    // ---- O += P * V : 16 c-tiles, B-fragments stream from LDS ----
    const bf16* vrow = Vcur + col;
#pragma unroll
    for (int t = 0; t < 16; ++t) {
      v16bf vb;
#pragma unroll
      for (int s = 0; s < 16; ++s)
        vb[s] = vrow[(size_t)((s < 8 ? s : s + 8) + halfOff) * VPITCH + t * 16];
      acc[t] = __builtin_amdgcn_wmma_f32_16x16x32_bf16(false, pa.v, false, vb,
                                                       (short)0, acc[t],
                                                       false, false);
    }

    // ---- close the pipeline stage: DMA must land before buffer swap ----
    if (more) wait_async0();
    __syncthreads();
    buf ^= 1;
  }

  // ---- epilogue: divide by softmax denominator, write out[b][c][m] ----
  // lacc[r] holds the denominator of row (m0 + r + halfOff) in every lane of
  // this lane's 16-group, so each lane divides its own columns directly.
#pragma unroll
  for (int r = 0; r < 8; ++r) {
    const float inv = 1.0f / lacc[r];
    const int m = m0 + r + halfOff;
#pragma unroll
    for (int t = 0; t < 16; ++t) {
      const int c = t * 16 + col;
      out[((size_t)b * C_ + c) * N_ + m] = acc[t][r] * inv;
    }
  }
}

// ---------------------------------------------------------------------------
extern "C" void kernel_launch(void* const* d_in, const int* in_sizes, int n_in,
                              void* d_out, int out_size, void* d_ws,
                              size_t ws_size, hipStream_t stream) {
  const float* x = (const float*)d_in[0];
  float* out = (float*)d_out;

  bf16* Xc = (bf16*)d_ws;                // [B,C,N] bf16, 16.8 MB
  bf16* Xb = Xc + (size_t)B_ * C_ * N_;  // [B,N,C] bf16, 16.8 MB

  const size_t total = (size_t)B_ * C_ * N_;
  convert_bf16_kernel<<<(int)(total / 4 / 256), 256, 0, stream>>>(x, Xc);
  transpose_bf16_kernel<<<dim3(N_ / 32, C_ / 32, B_), dim3(32, 8), 0, stream>>>(
      x, Xb);
  fa_kernel<<<dim3(N_ / 128, B_), 256, SMEM_BYTES, stream>>>(Xb, Xc, out);
}
